// VectorizedMAPGate_2705829397194
// MI455X (gfx1250) — matrix operand order repaired and checked by reading, not compiled, not run on hardware
//
#include <hip/hip_runtime.h>
#include <hip/hip_bf16.h>

// ---------------------------------------------------------------------------
// Problem constants (match reference): P experts, D=256, IN_DIM=2D=512, H=512
// ---------------------------------------------------------------------------
static constexpr int kP  = 16;
static constexpr int kD  = 256;
static constexpr int kIN = 512;   // 2*D
static constexpr int kH  = 512;
static constexpr int kMT  = 8;          // 16-token m-tiles register-blocked per wave
static constexpr int kTOK = kMT * 16;   // 128 tokens per block

typedef __attribute__((ext_vector_type(16))) __bf16 v16bf;
typedef __attribute__((ext_vector_type(8)))  float  v8f;

// float -> bf16 bits, round-to-nearest-even
static __device__ __forceinline__ unsigned short f2bf_u(float f) {
    union { float f; unsigned u; } c; c.f = f;
    unsigned r = c.u + 0x7FFFu + ((c.u >> 16) & 1u);
    return (unsigned short)(r >> 16);
}
static __device__ __forceinline__ __bf16 f2bf(float f) {
    unsigned short hs = f2bf_u(f);
    return __builtin_bit_cast(__bf16, hs);
}

// ---------------------------------------------------------------------------
// Prep: repack fc1_weight (P, IN_DIM, H) fp32 into bf16 B-fragments.
// Fragment chunk = one wave's B operand for one (p, ntile, kstep):
//   chunk[lane*16 + j] with j = 2*v + s holds W1[p][k][n],
//   n = ntile*16 + (lane&15), k = kstep*32 + 2*v + s + (lane>=16 ? 16 : 0)
// ---------------------------------------------------------------------------
__global__ __launch_bounds__(256)
void pack_w1_bf16(const float* __restrict__ w1, __bf16* __restrict__ packed) {
    int idx   = blockIdx.x * 256 + threadIdx.x;   // 0 .. 4,194,303
    int j     = idx & 15;
    int lane  = (idx >> 4) & 31;
    int kstep = (idx >> 9) & 15;
    int ntile = (idx >> 13) & 31;
    int p     = idx >> 18;
    int v = j >> 1, s = j & 1;
    int n = ntile * 16 + (lane & 15);
    int k = kstep * 32 + 2 * v + s + ((lane >> 4) ? 16 : 0);
    packed[idx] = f2bf(w1[((size_t)p * kIN + k) * kH + n]);
}

// ---------------------------------------------------------------------------
// Fused kernel: per block = (expert p, 128-token group).
//   Two K-phases: stage [128 tok x 256 k] bf16 in 64 KB LDS (phase0 = h,
//   phase1 = x), accumulators persist in registers across phases.
//   8 waves x 4 N-tiles x 8 M-tiles; each B fragment reused by 8 WMMAs.
//   Epilogue: in-lane nt-collapse, then one shfl-tree + ds_add per (mt,j).
// ---------------------------------------------------------------------------
__global__ __launch_bounds__(256)
void fused_expert_mlp(const float* __restrict__ hbuf,
                      const float* __restrict__ xbuf,
                      const __bf16* __restrict__ w1p,
                      const float* __restrict__ b1,
                      const float* __restrict__ w2,
                      const float* __restrict__ b2,
                      float* __restrict__ out) {
    // 64 KB LDS: activation tile during GEMM, fc2 accumulators afterwards
    __shared__ __align__(16) unsigned char smem[kTOK * kD * sizeof(__bf16)];
    __bf16* lds_a = reinterpret_cast<__bf16*>(smem);
    float*  accum = reinterpret_cast<float*>(smem);

    const int p     = blockIdx.x & (kP - 1);
    const int m0    = (blockIdx.x >> 4) * kTOK;
    const int t     = threadIdx.x;
    const int wave  = t >> 5;
    const int lane  = t & 31;
    const int lrow  = lane & 15;   // A: M row / B: N col
    const int lhalf = lane >> 4;

    v8f acc[kMT][4];
    #pragma unroll
    for (int mt = 0; mt < kMT; ++mt)
        #pragma unroll
        for (int nt = 0; nt < 4; ++nt)
            acc[mt][nt] = (v8f){};

    // Base of this wave's packed B fragments (ntile = wave*4 + nt, kstep = kkg)
    const __bf16* bb = w1p + ((size_t)(p * 32 + wave * 4) * 16) * 512 + lane * 16;

    for (int s = 0; s < 2; ++s) {
        if (s) __syncthreads();                 // all phase-0 A reads done
        const float* src = s ? xbuf : hbuf;     // concat split == phase split
        // Vectorized staging: float4 global loads -> packed bf16x4 LDS stores
        for (int q = t; q < kTOK * (kD / 4); q += 256) {
            int m  = q >> 6;                    // /64 float4s per row
            int f4 = q & 63;
            const float4 fv = *reinterpret_cast<const float4*>(
                src + ((size_t)(m0 + m) * kP + p) * kD + f4 * 4);
            uint2 pk;
            pk.x = (unsigned)f2bf_u(fv.x) | ((unsigned)f2bf_u(fv.y) << 16);
            pk.y = (unsigned)f2bf_u(fv.z) | ((unsigned)f2bf_u(fv.w) << 16);
            *reinterpret_cast<uint2*>(&lds_a[m * kD + f4 * 4]) = pk;
        }
        __syncthreads();

        for (int kk8 = 0; kk8 < 8; ++kk8) {
            const int kkg = s * 8 + kk8;        // global k-step 0..15
            v16bf bf[4];
            #pragma unroll
            for (int nt = 0; nt < 4; ++nt)
                bf[nt] = *reinterpret_cast<const v16bf*>(bb + ((size_t)nt * 16 + kkg) * 512);
            #pragma unroll
            for (int mt = 0; mt < kMT; ++mt) {
                // A fragment, contiguous k order (same as packed B): VGPR v -> 2v + 16*lhalf
                union { v16bf v; unsigned u[8]; } au;
                #pragma unroll
                for (int vv = 0; vv < 8; ++vv) {
                    int kl = 2 * vv + 16 * lhalf;  // even -> 4B aligned
                    au.u[vv] = *reinterpret_cast<const unsigned*>(
                        &lds_a[(mt * 16 + lrow) * kD + kk8 * 32 + kl]);
                }
                #pragma unroll
                for (int nt = 0; nt < 4; ++nt)
                    acc[mt][nt] = __builtin_amdgcn_wmma_f32_16x16x32_bf16(
                        false, au.v, false, bf[nt], (short)0, acc[mt][nt], false, false);
            }
        }
    }

    // Repurpose LDS as fc2 accumulators (all A reads complete)
    __syncthreads();
    if (t < kTOK) accum[t] = 0.0f;
    __syncthreads();

    // fc1 bias + fc2 weight for this lane's 4 columns
    float bias[4], w2v[4];
    #pragma unroll
    for (int nt = 0; nt < 4; ++nt) {
        int n = (wave * 4 + nt) * 16 + lrow;
        bias[nt] = b1[p * kH + n];
        w2v[nt]  = w2[p * kH + n];
    }

    #pragma unroll
    for (int mt = 0; mt < kMT; ++mt) {
        // In-lane collapse over the 4 N-tiles (same output rows, no shuffles)
        float part[8];
        #pragma unroll
        for (int j = 0; j < 8; ++j) part[j] = 0.0f;
        #pragma unroll
        for (int nt = 0; nt < 4; ++nt) {
            #pragma unroll
            for (int j = 0; j < 8; ++j) {
                float z = acc[mt][nt][j] + bias[nt];
                z = z > 0.0f ? z : 0.0f;
                part[j] = fmaf(z, w2v[nt], part[j]);
            }
        }
        // One shuffle tree + one LDS atomic per (mt, j)
        #pragma unroll
        for (int j = 0; j < 8; ++j) {
            float vp = part[j];
            vp += __shfl_xor(vp, 8, 32);   // xor stays within each 16-lane group
            vp += __shfl_xor(vp, 4, 32);
            vp += __shfl_xor(vp, 2, 32);
            vp += __shfl_xor(vp, 1, 32);
            if (lrow == 0) {
                int m = mt * 16 + j + 8 * lhalf;  // C layout: VGPR j -> M=j / j+8
                atomicAdd(&accum[m], vp);
            }
        }
    }
    __syncthreads();

    if (t < kTOK) {
        float zv = accum[t] + b2[p];
        out[(size_t)(m0 + t) * kP + p] = 1.0f / (1.0f + __expf(-zv));
    }
}

// ---------------------------------------------------------------------------
// Host entry
// ---------------------------------------------------------------------------
extern "C" void kernel_launch(void* const* d_in, const int* in_sizes, int n_in,
                              void* d_out, int out_size, void* d_ws, size_t ws_size,
                              hipStream_t stream) {
    const float* h  = (const float*)d_in[0];
    const float* x  = (const float*)d_in[1];
    const float* w1 = (const float*)d_in[2];
    const float* b1 = (const float*)d_in[3];
    const float* w2 = (const float*)d_in[4];
    const float* b2 = (const float*)d_in[5];
    float* out = (float*)d_out;

    __bf16* w1p = (__bf16*)d_ws;  // needs 16*512*512*2 B = 8 MiB

    const int packElems = kP * 32 * 16 * 512;            // 4,194,304
    pack_w1_bf16<<<packElems / 256, 256, 0, stream>>>(w1, w1p);

    const int ntok = in_sizes[0] / (kP * kD);            // B*T = 8192
    fused_expert_mlp<<<(ntok / kTOK) * kP, 256, 0, stream>>>(h, x, w1p, b1, w2, b2, out);
}
